// HFAX_CA_12876311953973
// MI455X (gfx1250) — compile-verified, matrix operand order checked by reading, not run on hardware
//
#include <hip/hip_runtime.h>
#include <hip/hip_bf16.h>
#include <math.h>

// ---------------------------------------------------------------------------
// CDNA5 (gfx1250) cross-window attention + fusion + MLP head
// wave32, WMMA f32_16x16x32_f16 for all GEMMs, flash-style online softmax.
// ---------------------------------------------------------------------------

typedef _Float16 v16h __attribute__((ext_vector_type(16)));
typedef float    v8f  __attribute__((ext_vector_type(8)));

#define WMMA_F16(a, b, c) \
  __builtin_amdgcn_wmma_f32_16x16x32_f16(false, (a), false, (b), (short)0, (c), false, false)

constexpr int   D     = 128;   // model dim == inner dim (4 heads x 32)
constexpr int   TQ    = 64;    // distinct query tokens per window (8x8)
constexpr int   TK    = 264;   // kv tokens per window (6 views x 4 x 11)
constexpr int   TKP   = 288;   // padded to 18 x 16
constexpr int   STR   = 132;   // f16 LDS row stride (66 dwords, bank-friendly)
constexpr int   STRT  = 296;   // transposed-vh row stride (148 dwords)
constexpr int   PS    = 40;    // P-stage row stride (halves)
constexpr float ATT_SCALE = 0.17677669529663689f;  // 32^-0.5

// dynamic LDS layout (bytes)
constexpr int OFF_B1   = 0;                       // 288x132 f16: kn / vn / qn / ah
constexpr int OFF_B2   = OFF_B1 + TKP * STR * 2;  // kh  (tokens x channels)
constexpr int OFF_B3T  = OFF_B2 + TKP * STR * 2;  // vh^T (channels x tokens)
constexpr int OFF_B4   = OFF_B3T + D * STRT * 2;  // qh (64x132)
constexpr int OFF_PST  = OFF_B4 + TQ * STR * 2;   // 8 waves x 32 x PS f16
constexpr int OFF_MEAN = OFF_PST + 8 * 32 * PS * 2;
constexpr int OFF_RSTD = OFF_MEAN + TK * 4;
constexpr int SMEM_BYTES = OFF_RSTD + TK * 4;
static_assert(SMEM_BYTES <= 320 * 1024, "LDS budget (320KB/WGP)");
static_assert((OFF_B3T & 15) == 0 && ((STRT * 2) & 15) == 0, "b128 alignment");

struct ScaleArgs {
  const float* x;    // (d, H, W) BEV features
  const float* img;  // (n=6, d, h, w) camera features
  const float *ln_q_g, *ln_q_b, *ln_k_g, *ln_k_b, *ln_v_g, *ln_v_b;
  const float *wq, *bq, *wk, *bk, *wv, *bv, *wo, *bo;
  float* out;        // (H*W, d) channels-last attention output
  int H, W, h, w, X, Y, L;
};

// ---- fragment helpers (documented CDNA5 16-bit WMMA VGPR layouts) ----------
// A (16x32, row-major MxK) and B sourced from row-major NxK ("transposed B")
// share the per-lane gather: row = lane&15, kbase = (lane>=16) ? 8 : 0,
// halves i<8 -> K=kbase+i, i>=8 -> K=kbase+16+(i-8).  Two ds_load_b128 each.
__device__ __forceinline__ v16h frag_ld_trans(const _Float16* base, int row0,
                                              int col0, int stride) {
  const int lane = threadIdx.x & 31;
  const _Float16* p = base + (size_t)(row0 + (lane & 15)) * stride + col0 +
                      ((lane >> 4) ? 8 : 0);
  v16h r;
#pragma unroll
  for (int i = 0; i < 8; ++i) { r[i] = p[i]; r[i + 8] = p[16 + i]; }
  return r;
}

// B from torch-style fp32 weight W(N,K=128) row-major: B[k][n] = W[n][k]
__device__ __forceinline__ v16h frag_ld_w(const float* W, int n0, int k0) {
  const int lane = threadIdx.x & 31;
  const float* p = W + (size_t)(n0 + (lane & 15)) * 128 + k0 + ((lane >> 4) ? 8 : 0);
  v16h r;
#pragma unroll
  for (int i = 0; i < 8; ++i) {
    r[i]     = (_Float16)p[i];
    r[i + 8] = (_Float16)p[16 + i];
  }
  return r;
}

// store f32 D fragment as f16, row-major dst: row = v + 8*(lane>>4), col = lane&15
__device__ __forceinline__ void frag_st(v8f d, _Float16* base, int row0,
                                        int col0, int stride) {
  const int lane = threadIdx.x & 31;
  _Float16* p = base + (size_t)(row0 + ((lane >> 4) ? 8 : 0)) * stride + col0 +
                (lane & 15);
#pragma unroll
  for (int v = 0; v < 8; ++v) p[(size_t)v * stride] = (_Float16)d[v];
}

// store f32 D fragment as f16 TRANSPOSED: dstT[col][row]. Per lane the 8
// row-values are contiguous -> single 16B-aligned ds_store_b128.
__device__ __forceinline__ void frag_st_T(v8f d, _Float16* baseT, int row0,
                                          int col0, int strideT) {
  const int lane = threadIdx.x & 31;
  _Float16* p = baseT + (size_t)(col0 + (lane & 15)) * strideT + row0 +
                ((lane >> 4) ? 8 : 0);
#pragma unroll
  for (int v = 0; v < 8; ++v) p[v] = (_Float16)d[v];
}

// tiled GEMM: Dst(M x 128) = A(M x 128) @ W^T + bias, M = mtiles*16
__device__ __forceinline__ void proj_gemm(const _Float16* A, int mtiles,
                                          const float* W, const float* bias,
                                          _Float16* Dst, int wave,
                                          bool transposed) {
  const int total = mtiles * 8;
  __builtin_prefetch(W, 0, 3);  // global_prefetch_b8
  for (int tile = wave; tile < total; tile += 8) {
    const int mt = tile >> 3, nt = tile & 7;
    v8f acc = {};
#pragma unroll
    for (int kt = 0; kt < 4; ++kt) {
      v16h a = frag_ld_trans(A, mt * 16, kt * 32, STR);
      v16h b = frag_ld_w(W, nt * 16, kt * 32);
      acc = WMMA_F16(a, b, acc);
    }
    const float bv = bias[nt * 16 + (threadIdx.x & 15)];
#pragma unroll
    for (int v = 0; v < 8; ++v) acc[v] += bv;
    if (transposed) frag_st_T(acc, Dst, mt * 16, nt * 16, STRT);
    else            frag_st(acc, Dst, mt * 16, nt * 16, STR);
  }
}

// online softmax update for one 16x16 score tile (D layout)
__device__ __forceinline__ void softmax_update(v8f S, bool colInvalid,
                                               float* mrow, float* lrow,
                                               v8f& oA, v8f& oB,
                                               _Float16* pst, int mt, int sub) {
  const int lane = threadIdx.x & 31;
  const int half = lane >> 4;
  const int n = lane & 15;
#pragma unroll
  for (int v = 0; v < 8; ++v) {
    float sv = colInvalid ? -1.0e30f : (float)S[v] * ATT_SCALE;
    float rmax = sv;
    rmax = fmaxf(rmax, __shfl_xor(rmax, 1, 16));
    rmax = fmaxf(rmax, __shfl_xor(rmax, 2, 16));
    rmax = fmaxf(rmax, __shfl_xor(rmax, 4, 16));
    rmax = fmaxf(rmax, __shfl_xor(rmax, 8, 16));
    const float nm = fmaxf(mrow[v], rmax);
    const float corr = __expf(mrow[v] - nm);
    mrow[v] = nm;
    const float p = __expf(sv - nm);
    float rsum = p;
    rsum += __shfl_xor(rsum, 1, 16);
    rsum += __shfl_xor(rsum, 2, 16);
    rsum += __shfl_xor(rsum, 4, 16);
    rsum += __shfl_xor(rsum, 8, 16);
    lrow[v] = lrow[v] * corr + rsum;
    oA[v] *= corr;
    oB[v] *= corr;
    pst[(size_t)(mt * 16 + v + half * 8) * PS + sub * 16 + n] = (_Float16)p;
  }
}

// ---------------------------------------------------------------------------
__global__ __launch_bounds__(256) void attn_kernel(ScaleArgs s0, ScaleArgs s1,
                                                   int L0) {
  const bool second = (int)blockIdx.x >= L0;
  const ScaleArgs S = second ? s1 : s0;
  const int l = (int)blockIdx.x - (second ? L0 : 0);
  const int wx = l / S.Y, wy = l % S.Y;
  const int x0 = wx * 8, y0 = wy * 8;      // query window origin
  const int gr0 = wx * 4, gc0 = wy * 11;   // kv window origin
  const int HW = S.H * S.W;
  const int hw = S.h * S.w;
  const int tid = threadIdx.x;
  // scalarize: wave id is uniform across the wave -> keep loops in SALU,
  // leave EXEC untouched around WMMA.
  const int wave = __builtin_amdgcn_readfirstlane(tid >> 5);
  const int lane = tid & 31;

  extern __shared__ char smem[];
  _Float16* B1   = (_Float16*)(smem + OFF_B1);
  _Float16* B2   = (_Float16*)(smem + OFF_B2);
  _Float16* B3T  = (_Float16*)(smem + OFF_B3T);
  _Float16* B4   = (_Float16*)(smem + OFF_B4);
  _Float16* PST  = (_Float16*)(smem + OFF_PST);
  float*    MEAN = (float*)(smem + OFF_MEAN);
  float*    RSTD = (float*)(smem + OFF_RSTD);

  // ---- phase 1: kv LN statistics -----------------------------------------
  for (int t = tid; t < TK; t += 256) {
    const int n = t / 44, rr = (t % 44) / 11, cq = t % 11;
    const float* p = S.img + (size_t)n * D * hw + (size_t)(gr0 + rr) * S.w + (gc0 + cq);
    float s = 0.f, s2 = 0.f;
    for (int c = 0; c < D; ++c) {
      const float v = p[(size_t)c * hw];
      s += v; s2 += v * v;
    }
    const float m = s * (1.0f / D);
    MEAN[t] = m;
    RSTD[t] = rsqrtf(s2 * (1.0f / D) - m * m + 1e-5f);
  }
  __syncthreads();

  // ---- phase 2/3: K normalize -> kh, phase 4/5: V normalize -> vh^T ------
  for (int pass = 0; pass < 2; ++pass) {
    const float* g  = pass ? S.ln_v_g : S.ln_k_g;
    const float* bb = pass ? S.ln_v_b : S.ln_k_b;
    // paired channels -> merged ds_store_b32
    for (int idx = tid; idx < TKP * (D / 2); idx += 256) {
      const int t = idx % TKP;
      const int c = (idx / TKP) * 2;
      _Float16 v0 = (_Float16)0.f, v1 = (_Float16)0.f;
      if (t < TK) {
        const int n = t / 44, rr = (t % 44) / 11, cq = t % 11;
        const size_t base = (size_t)n * D * hw + (size_t)(gr0 + rr) * S.w + (gc0 + cq);
        const float r0 = S.img[base + (size_t)c * hw];
        const float r1 = S.img[base + (size_t)(c + 1) * hw];
        const float mn = MEAN[t], rs = RSTD[t];
        v0 = (_Float16)((r0 - mn) * rs * g[c] + bb[c]);
        v1 = (_Float16)((r1 - mn) * rs * g[c + 1] + bb[c + 1]);
      }
      _Float16* p = B1 + (size_t)t * STR + c;
      p[0] = v0; p[1] = v1;
    }
    __syncthreads();
    if (pass == 0) proj_gemm(B1, TKP / 16, S.wk, S.bk, B2, wave, false);
    else           proj_gemm(B1, TKP / 16, S.wv, S.bv, B3T, wave, true);
    __syncthreads();
  }

  // ---- phase 6: Q statistics + normalize (overwrites B1) -----------------
  for (int t = tid; t < TQ; t += 256) {
    const int i = t >> 3, j = t & 7;
    const float* p = S.x + (size_t)(x0 + i) * S.W + (y0 + j);
    float s = 0.f, s2 = 0.f;
    for (int c = 0; c < D; ++c) {
      const float v = p[(size_t)c * HW];
      s += v; s2 += v * v;
    }
    const float m = s * (1.0f / D);
    MEAN[t] = m;
    RSTD[t] = rsqrtf(s2 * (1.0f / D) - m * m + 1e-5f);
  }
  __syncthreads();
  for (int idx = tid; idx < TQ * (D / 2); idx += 256) {
    const int t = idx % TQ;
    const int c = (idx / TQ) * 2;
    const int i = t >> 3, j = t & 7;
    const size_t base = (size_t)(x0 + i) * S.W + (y0 + j);
    const float r0 = S.x[(size_t)c * HW + base];
    const float r1 = S.x[(size_t)(c + 1) * HW + base];
    const float mn = MEAN[t], rs = RSTD[t];
    _Float16* p = B1 + (size_t)t * STR + c;
    p[0] = (_Float16)((r0 - mn) * rs * S.ln_q_g[c] + S.ln_q_b[c]);
    p[1] = (_Float16)((r1 - mn) * rs * S.ln_q_g[c + 1] + S.ln_q_b[c + 1]);
  }
  __syncthreads();

  // ---- phase 7: qh = qn @ Wq^T + bq --------------------------------------
  proj_gemm(B1, TQ / 16, S.wq, S.bq, B4, wave, false);
  __syncthreads();

  // ---- phase 8: flash attention per wave ---------------------------------
  // head = wave/2, row slab r0 = (wave&1)*32 (two 16-row M tiles)
  {
    const int head = wave >> 1;
    const int hd0 = head * 32;
    const int r0 = (wave & 1) * 32;
    _Float16* pst = PST + (size_t)wave * 32 * PS;

    const v16h qa0 = frag_ld_trans(B4, r0, hd0, STR);
    const v16h qa1 = frag_ld_trans(B4, r0 + 16, hd0, STR);
    v8f oacc[2][2] = {{{}, {}}, {{}, {}}};
    float mrun[2][8], lrun[2][8];
#pragma unroll
    for (int mt = 0; mt < 2; ++mt)
#pragma unroll
      for (int v = 0; v < 8; ++v) { mrun[mt][v] = -1.0e30f; lrun[mt][v] = 0.f; }

    for (int kp = 0; kp < TKP / 32; ++kp) {
#pragma unroll
      for (int sub = 0; sub < 2; ++sub) {
        const int tok0 = kp * 32 + sub * 16;
        const v16h kb = frag_ld_trans(B2, tok0, hd0, STR);  // K^T fragment
        v8f sc0 = WMMA_F16(qa0, kb, v8f{});
        v8f sc1 = WMMA_F16(qa1, kb, v8f{});
        const bool inval = (tok0 + (lane & 15)) >= TK;
        softmax_update(sc0, inval, mrun[0], lrun[0], oacc[0][0], oacc[0][1], pst, 0, sub);
        softmax_update(sc1, inval, mrun[1], lrun[1], oacc[1][0], oacc[1][1], pst, 1, sub);
      }
      asm volatile("s_wait_dscnt 0" ::: "memory");  // P staged in LDS (same wave)
      const v16h pa0 = frag_ld_trans(pst, 0, 0, PS);
      const v16h pa1 = frag_ld_trans(pst, 16, 0, PS);
#pragma unroll
      for (int nt = 0; nt < 2; ++nt) {
        // B[k][n] = V[tok k][ch n] = vh^T[ch][tok] -> cheap trans-pattern load
        const v16h bv = frag_ld_trans(B3T, hd0 + nt * 16, kp * 32, STRT);
        oacc[0][nt] = WMMA_F16(pa0, bv, oacc[0][nt]);
        oacc[1][nt] = WMMA_F16(pa1, bv, oacc[1][nt]);
      }
    }
    // epilogue: normalize, write ah (overwrites B1 query region)
    const int half = lane >> 4;
#pragma unroll
    for (int mt = 0; mt < 2; ++mt)
#pragma unroll
      for (int nt = 0; nt < 2; ++nt) {
        _Float16* p = B1 + (size_t)(r0 + mt * 16 + half * 8) * STR + hd0 +
                      nt * 16 + (lane & 15);
#pragma unroll
        for (int v = 0; v < 8; ++v)
          p[(size_t)v * STR] = (_Float16)(oacc[mt][nt][v] / lrun[mt][v]);
      }
  }
  __syncthreads();

  // ---- phase 9: z = ah @ Wo^T + bo + skip, store channels-last -----------
  __builtin_prefetch(S.wo, 0, 3);
  for (int tile = wave; tile < 32; tile += 8) {
    const int mt = tile >> 3, nt = tile & 7;
    v8f acc = {};
#pragma unroll
    for (int kt = 0; kt < 4; ++kt) {
      v16h a = frag_ld_trans(B1, mt * 16, kt * 32, STR);
      v16h b = frag_ld_w(S.wo, nt * 16, kt * 32);
      acc = WMMA_F16(a, b, acc);
    }
    const int o = nt * 16 + (lane & 15);
    const float bo_v = S.bo[o];
    const int half = lane >> 4;
#pragma unroll
    for (int v = 0; v < 8; ++v) {
      const int t = mt * 16 + v + half * 8;
      const int pi = t >> 3, pj = t & 7;
      const size_t pix = (size_t)(x0 + pi) * S.W + (y0 + pj);
      const float skip = S.x[(size_t)o * HW + pix];
      S.out[pix * D + o] = acc[v] + bo_v + skip;
    }
  }
}

// ---------------------------------------------------------------------------
// fuse: out = Linear(LN(cat[a, bilinear(other)])) + a   (channels-last)
// grid = H*W blocks, 128 threads
__global__ __launch_bounds__(128) void fuse_kernel(
    const float* __restrict__ a, int H, int W, const float* __restrict__ other,
    int oh, int ow, const float* __restrict__ g, const float* __restrict__ bvec,
    const float* __restrict__ w, const float* __restrict__ bias,
    float* __restrict__ out) {
  __shared__ float cat[256], red[128], normed[256];
  const int pix = blockIdx.x;
  const int py = pix / W, px = pix % W;
  const int o = threadIdx.x;

  const float aval = a[(size_t)pix * 128 + o];
  // bilinear, align_corners=True
  const float posr = py * ((float)(oh - 1) / (float)(H - 1));
  const float posc = px * ((float)(ow - 1) / (float)(W - 1));
  int r0i = (int)posr; r0i = r0i > oh - 2 ? oh - 2 : r0i;
  int c0i = (int)posc; c0i = c0i > ow - 2 ? ow - 2 : c0i;
  const float fr = posr - r0i, fc = posc - c0i;
  const float* p00 = other + ((size_t)r0i * ow + c0i) * 128;
  const float biv = (1.f - fr) * (1.f - fc) * p00[o] +
                    (1.f - fr) * fc * p00[128 + o] +
                    fr * (1.f - fc) * p00[(size_t)ow * 128 + o] +
                    fr * fc * p00[(size_t)ow * 128 + 128 + o];
  cat[o] = aval;
  cat[128 + o] = biv;
  __syncthreads();

  red[o] = cat[o] + cat[o + 128];
  __syncthreads();
  for (int s = 64; s > 0; s >>= 1) {
    if (o < s) red[o] += red[o + s];
    __syncthreads();
  }
  const float mean = red[0] * (1.0f / 256.0f);
  __syncthreads();
  const float d0 = cat[o] - mean, d1 = cat[o + 128] - mean;
  red[o] = d0 * d0 + d1 * d1;
  __syncthreads();
  for (int s = 64; s > 0; s >>= 1) {
    if (o < s) red[o] += red[o + s];
    __syncthreads();
  }
  const float rstd = rsqrtf(red[0] * (1.0f / 256.0f) + 1e-5f);
  __syncthreads();
  normed[o]       = d0 * rstd * g[o] + bvec[o];
  normed[o + 128] = d1 * rstd * g[o + 128] + bvec[o + 128];
  __syncthreads();

  float acc = bias[o];
  const float* wr = w + (size_t)o * 256;
#pragma unroll 4
  for (int c = 0; c < 256; ++c) acc += normed[c] * wr[c];
  out[(size_t)pix * 128 + o] = acc + aval;
}

// ---------------------------------------------------------------------------
// head: out = x + W2 @ gelu(W1 @ LN(x) + b1) + b2   (channels-last)
// grid = H*W blocks, 256 threads
__global__ __launch_bounds__(256) void head_kernel(
    const float* __restrict__ in, const float* __restrict__ png,
    const float* __restrict__ pnb, const float* __restrict__ w1,
    const float* __restrict__ b1, const float* __restrict__ w2,
    const float* __restrict__ b2, float* __restrict__ out) {
  __shared__ float tv[128], normed[128], gv[256], red[256];
  const int pix = blockIdx.x;
  const int tid = threadIdx.x;

  if (tid < 128) tv[tid] = in[(size_t)pix * 128 + tid];
  __syncthreads();
  red[tid] = (tid < 128) ? tv[tid] : 0.f;
  __syncthreads();
  for (int s = 128; s > 0; s >>= 1) {
    if (tid < s) red[tid] += red[tid + s];
    __syncthreads();
  }
  const float mean = red[0] * (1.0f / 128.0f);
  __syncthreads();
  const float dv = (tid < 128) ? (tv[tid] - mean) : 0.f;
  red[tid] = dv * dv;
  __syncthreads();
  for (int s = 128; s > 0; s >>= 1) {
    if (tid < s) red[tid] += red[tid + s];
    __syncthreads();
  }
  const float rstd = rsqrtf(red[0] * (1.0f / 128.0f) + 1e-5f);
  __syncthreads();
  if (tid < 128) normed[tid] = dv * rstd * png[tid] + pnb[tid];
  __syncthreads();

  float hsum = b1[tid];
  const float* w1r = w1 + (size_t)tid * 128;
#pragma unroll 4
  for (int c = 0; c < 128; ++c) hsum += normed[c] * w1r[c];
  gv[tid] = 0.5f * hsum * (1.f + erff(hsum * 0.70710678118654752f));  // exact gelu
  __syncthreads();

  if (tid < 128) {
    float acc = b2[tid];
    const float* w2r = w2 + (size_t)tid * 256;
#pragma unroll 4
    for (int c = 0; c < 256; ++c) acc += gv[c] * w2r[c];
    out[(size_t)pix * 128 + tid] = tv[tid] + acc;
  }
}

// ---------------------------------------------------------------------------
extern "C" void kernel_launch(void* const* d_in, const int* in_sizes, int n_in,
                              void* d_out, int out_size, void* d_ws,
                              size_t ws_size, hipStream_t stream) {
  auto P = [&](int i) { return (const float*)d_in[i]; };
  const float* x1 = P(0);
  const float* x2 = P(1);
  const float* img1 = P(2);
  const float* img2 = P(3);
  // params flattened in setup_inputs insertion order:
  // s1: 4..17  s2: 18..31  proj_1_2: 32..35  proj_2_1: 36..39  head: 40..45

  float* ws = (float*)d_ws;
  float* A1 = ws;                 // 128*128*128 = 2,097,152 (attn out, cl)
  float* A2 = A1 + 2097152;       // 64*64*128   =   524,288
  float* F1 = A2 + 524288;        // fused scale-1
  float* F2 = F1 + 2097152;       // fused scale-2

  ScaleArgs s0, s1;
  s0.x = x1; s0.img = img1;
  s0.ln_q_g = P(4);  s0.ln_q_b = P(5);  s0.ln_k_g = P(6);  s0.ln_k_b = P(7);
  s0.ln_v_g = P(8);  s0.ln_v_b = P(9);
  s0.wq = P(10); s0.bq = P(11); s0.wk = P(12); s0.bk = P(13);
  s0.wv = P(14); s0.bv = P(15); s0.wo = P(16); s0.bo = P(17);
  s0.out = A1; s0.H = 128; s0.W = 128; s0.h = 64; s0.w = 176;
  s0.X = 16; s0.Y = 16; s0.L = 256;

  s1.x = x2; s1.img = img2;
  s1.ln_q_g = P(18); s1.ln_q_b = P(19); s1.ln_k_g = P(20); s1.ln_k_b = P(21);
  s1.ln_v_g = P(22); s1.ln_v_b = P(23);
  s1.wq = P(24); s1.bq = P(25); s1.wk = P(26); s1.bk = P(27);
  s1.wv = P(28); s1.bv = P(29); s1.wo = P(30); s1.bo = P(31);
  s1.out = A2; s1.H = 64; s1.W = 64; s1.h = 32; s1.w = 88;
  s1.X = 8; s1.Y = 8; s1.L = 64;

  // one workgroup per window; scale-1 first (256 blocks), then scale-2 (64)
  attn_kernel<<<s0.L + s1.L, 256, SMEM_BYTES, stream>>>(s0, s1, s0.L);

  // sequential cross-scale fusion (o1 updated first, then feeds o2's fuse)
  fuse_kernel<<<128 * 128, 128, 0, stream>>>(A1, 128, 128, A2, 64, 64,
                                             P(32), P(33), P(34), P(35), F1);
  fuse_kernel<<<64 * 64, 128, 0, stream>>>(A2, 64, 64, F1, 128, 128,
                                           P(36), P(37), P(38), P(39), F2);

  float* out_f = (float*)d_out;
  head_kernel<<<128 * 128, 256, 0, stream>>>(F1, P(40), P(41), P(42), P(43),
                                             P(44), P(45), out_f);
  head_kernel<<<64 * 64, 256, 0, stream>>>(F2, P(40), P(41), P(42), P(43),
                                           P(44), P(45), out_f + 2097152);
}
